// HierNet_3092376453580
// MI455X (gfx1250) — matrix operand order repaired and checked by reading, not compiled, not run on hardware
//
#include <hip/hip_runtime.h>

// ---------------------------------------------------------------------------
// HierNet forward for MI455X (gfx1250, wave32, WMMA).
//   Stage 1: nout[n,b,k] = x @ W[n] + bias   -> bf16 WMMA 16x16x32, f32 acc
//            each wave owns a 3-node x 2-batch-tile macro tile:
//            per K-step 10 b128 loads feed 6 WMMAs (reg-level A/B reuse)
//   Stage 2: codewords collapsed to <=8 sparse (n,k,coef) per class (table)
//   Stage 3: gates / clip / mask / softmax-base, one block per batch row
// ---------------------------------------------------------------------------

typedef __attribute__((ext_vector_type(16))) __bf16 bf16x16;
typedef __attribute__((ext_vector_type(8)))  __bf16 bf16x8;
typedef __attribute__((ext_vector_type(8)))  float  f32x8;

#define B_ 512
#define D_ 2048
#define N_ 111
#define K_ 10
#define C_ 1000
#define KT_ 64   // D_/32 K-steps per GEMM
#define BT_ 32   // B_/16 batch tiles

// ---- fp32 -> bf16 convert of x (row-major) --------------------------------
__global__ void k_cvt_x(const float* __restrict__ x, __bf16* __restrict__ xb, int n) {
    int i = blockIdx.x * blockDim.x + threadIdx.x;
    if (i < n) xb[i] = (__bf16)x[i];
}

// ---- pack W[111,2048,10] -> bf16 B-matrix lane layout, K padded to 16 -----
// WB[((n*64 + kt)*32 + lane)*16 + j]:
//   col = lane%16 (fanout k, zero for col>=10)
//   K-row d = kt*32 + (lane<16 ? 0 : 8) + (j&7) + (j<8 ? 0 : 16)   (j = 0..15)
__global__ void k_pack_w(const float* __restrict__ W, __bf16* __restrict__ WB) {
    int e = blockIdx.x * blockDim.x + threadIdx.x;
    if (e >= N_ * KT_ * 32 * 16) return;
    int j    = e & 15;
    int lane = (e >> 4) & 31;
    int kt   = (e >> 9) & 63;
    int n    = e >> 15;
    int d    = kt * 32 + ((lane & 16) ? 8 : 0) + (j & 7) + ((j & 8) ? 16 : 0);
    int col  = lane & 15;
    float v  = (col < K_) ? W[(n * D_ + d) * K_ + col] : 0.0f;
    WB[e] = (__bf16)v;
}

// ---- logm[n-1,c] = log(clip(masks[n,c], 1e-17, 1)) for n=1..110 -----------
__global__ void k_logm(const float* __restrict__ masks, float* __restrict__ logm) {
    int i = blockIdx.x * blockDim.x + threadIdx.x;
    if (i >= (N_ - 1) * C_) return;
    int n = i / C_ + 1;
    int c = i - (n - 1) * C_;
    float m = fminf(fmaxf(masks[n * C_ + c], 1e-17f), 1.0f);
    logm[i] = logf(m);
}

// ---- per-class sparse table from codewords (general, <=8 entries) ---------
__global__ void k_table(const float* __restrict__ cw, int* __restrict__ tidx,
                        float* __restrict__ tcoef, int* __restrict__ tcnt) {
    int c = blockIdx.x * blockDim.x + threadIdx.x;
    if (c >= C_) return;
    int cnt = 0;
    for (int n = 0; n < N_; ++n)
        for (int k = 0; k < K_; ++k) {
            float v = cw[(n * K_ + k) * C_ + c];
            if (v != 0.0f && cnt < 8) {
                tidx[c * 8 + cnt]  = (n << 4) | k;
                tcoef[c * 8 + cnt] = v;
                ++cnt;
            }
        }
    tcnt[c] = cnt;
}

// ---- batched GEMM: one wave per (3 nodes, 2 batch tiles) macro tile -------
// grid = 37*16 single-wave blocks (barrier-free), 6 WMMAs per K-step.
__global__ void __launch_bounds__(32)
k_gemm(const __bf16* __restrict__ xb, const __bf16* __restrict__ WB,
       const float* __restrict__ bias, float* __restrict__ nout) {
    const int wave = blockIdx.x;     // 0..591
    const int ng   = wave >> 4;      // node triple 0..36  (nodes 3ng..3ng+2)
    const int bp   = wave & 15;      // batch-tile pair    (tiles 2bp, 2bp+1)
    const int lane = threadIdx.x & 31;
    const int m    = lane & 15;      // A row within tile / C column id
    const int hi   = lane >> 4;      // lane half selects K/M sub-block

    // A (16x32 bf16): lane m, K-halves {0..7,16..23} (lo lanes) / {8..15,24..31}
    const __bf16* pa0 = xb + (bp * 32 + m) * D_ + hi * 8;  // batch tile 2bp
    const __bf16* pa1 = pa0 + 16 * D_;                     // batch tile 2bp+1
    // B (32x16 bf16): pre-packed contiguous 16 bf16 per lane per K-step
    const __bf16* pb0 = WB + (((ng * 3 + 0) * KT_) * 32 + lane) * 16;
    const __bf16* pb1 = pb0 + KT_ * 32 * 16;
    const __bf16* pb2 = pb1 + KT_ * 32 * 16;

    f32x8 c00 = {}, c01 = {}, c10 = {}, c11 = {}, c20 = {}, c21 = {};
    for (int kt = 0; kt < KT_; ++kt) {
        bf16x8 a0lo = *(const bf16x8*)(pa0);
        bf16x8 a0hi = *(const bf16x8*)(pa0 + 16);
        bf16x8 a1lo = *(const bf16x8*)(pa1);
        bf16x8 a1hi = *(const bf16x8*)(pa1 + 16);
        bf16x16 a0, a1;
        #pragma unroll
        for (int i = 0; i < 8; ++i) {
            a0[i] = a0lo[i]; a0[8 + i] = a0hi[i];
            a1[i] = a1lo[i]; a1[8 + i] = a1hi[i];
        }
        bf16x16 b0 = *(const bf16x16*)(pb0);
        bf16x16 b1 = *(const bf16x16*)(pb1);
        bf16x16 b2 = *(const bf16x16*)(pb2);
        c00 = __builtin_amdgcn_wmma_f32_16x16x32_bf16(false, a0, false, b0, (short)0, c00, false, false);
        c01 = __builtin_amdgcn_wmma_f32_16x16x32_bf16(false, a1, false, b0, (short)0, c01, false, false);
        c10 = __builtin_amdgcn_wmma_f32_16x16x32_bf16(false, a0, false, b1, (short)0, c10, false, false);
        c11 = __builtin_amdgcn_wmma_f32_16x16x32_bf16(false, a1, false, b1, (short)0, c11, false, false);
        c20 = __builtin_amdgcn_wmma_f32_16x16x32_bf16(false, a0, false, b2, (short)0, c20, false, false);
        c21 = __builtin_amdgcn_wmma_f32_16x16x32_bf16(false, a1, false, b2, (short)0, c21, false, false);
        pa0 += 32; pa1 += 32;
        pb0 += 32 * 16; pb1 += 32 * 16; pb2 += 32 * 16;
    }

    // C layout: VGPR r -> M = r (lanes 0-15) / r+8 (lanes 16-31), N = lane%16
    const int col = lane & 15;
    if (col < K_) {
#define STORE_TILE(CC, NODE, BT)                                          \
        {                                                                 \
            float bv = bias[(NODE) * K_ + col];                           \
            _Pragma("unroll")                                             \
            for (int r = 0; r < 8; ++r) {                                 \
                int gb = (BT) * 16 + (hi ? r + 8 : r);                    \
                nout[(NODE) * (B_ * K_) + gb * K_ + col] = CC[r] + bv;    \
            }                                                             \
        }
        const int n0 = ng * 3, t0 = bp * 2;
        STORE_TILE(c00, n0 + 0, t0 + 0)
        STORE_TILE(c01, n0 + 0, t0 + 1)
        STORE_TILE(c10, n0 + 1, t0 + 0)
        STORE_TILE(c11, n0 + 1, t0 + 1)
        STORE_TILE(c20, n0 + 2, t0 + 0)
        STORE_TILE(c21, n0 + 2, t0 + 1)
#undef STORE_TILE
    }
}

// ---- gates + clipped class logits + mask + softmax base -------------------
__global__ void __launch_bounds__(256)
k_final(const float* __restrict__ gate, const int* __restrict__ anc,
        const float* __restrict__ nout, const int* __restrict__ tidx,
        const float* __restrict__ tcoef, const int* __restrict__ tcnt,
        const float* __restrict__ logm, float* __restrict__ output,
        float* __restrict__ sfmx) {
    __shared__ float sg[N_];
    __shared__ float sw[N_];
    __shared__ float red[256];
    const int b = blockIdx.x;
    const int t = threadIdx.x;

    if (t < N_) sg[t] = gate[b * N_ + t];
    __syncthreads();
    if (t < N_) {  // w[n] = gate[n] * prod_{ancestors p} gate[p]
        float p = 1.0f;
        const int* ar = anc + t * N_;
        for (int j = 0; j < N_; ++j) p = ar[j] ? p * sg[j] : p;
        sw[t] = p * sg[t];
    }
    __syncthreads();

    float lsum = 0.0f;
    for (int c = t; c < C_; c += 256) {
        float accv = 0.0f;
        const int cnt = tcnt[c];
        for (int e = 0; e < cnt; ++e) {
            int pk = tidx[c * 8 + e];
            int n = pk >> 4, k = pk & 15;
            accv += tcoef[c * 8 + e] * sw[n] * nout[n * (B_ * K_) + b * K_ + k];
        }
        float out = fminf(fmaxf(accv, -1e17f), 10.0f);
        output[b * C_ + c] = out;
        // msum terms are all <= 0, so the ==0 test is order-independent/exact
        float msum = 0.0f;
        for (int n = 1; n < N_; ++n)
            msum = fmaf(logm[(n - 1) * C_ + c], 1.0f - sg[n], msum);
        if (msum == 0.0f) lsum += expf(out);
    }
    red[t] = lsum;
    __syncthreads();
    for (int s = 128; s > 0; s >>= 1) {
        if (t < s) red[t] += red[t + s];
        __syncthreads();
    }
    if (t == 0) sfmx[b] = red[0];
}

// ---------------------------------------------------------------------------
extern "C" void kernel_launch(void* const* d_in, const int* in_sizes, int n_in,
                              void* d_out, int out_size, void* d_ws, size_t ws_size,
                              hipStream_t stream) {
    const float* x     = (const float*)d_in[0];   // [512,2048]
    const float* gate  = (const float*)d_in[1];   // [512,111]
    const float* W     = (const float*)d_in[2];   // [111,2048,10]
    const float* bias  = (const float*)d_in[3];   // [111,10]
    const float* cw    = (const float*)d_in[4];   // [111,10,1000]
    const float* masks = (const float*)d_in[5];   // [111,1000]
    const int*   anc   = (const int*)d_in[6];     // [111,111]

    float* output = (float*)d_out;                 // [512,1000]
    float* nout   = output + B_ * C_;              // [111,512,10]
    float* sfmx   = nout + N_ * B_ * K_;           // [512]

    char* ws = (char*)d_ws;
    size_t off = 0;
    __bf16* xb = (__bf16*)(ws + off);  off += (size_t)B_ * D_ * 2;            // 2.0 MB
    __bf16* WB = (__bf16*)(ws + off);  off += (size_t)N_ * KT_ * 32 * 16 * 2; // 7.3 MB
    float* logm = (float*)(ws + off);  off += (size_t)(N_ - 1) * C_ * 4;      // 0.44 MB
    int*   tidx = (int*)(ws + off);    off += (size_t)C_ * 8 * 4;
    float* tcoef = (float*)(ws + off); off += (size_t)C_ * 8 * 4;
    int*   tcnt = (int*)(ws + off);    off += (size_t)C_ * 4;

    k_cvt_x<<<(B_ * D_ + 255) / 256, 256, 0, stream>>>(x, xb, B_ * D_);
    k_pack_w<<<(N_ * KT_ * 32 * 16) / 256, 256, 0, stream>>>(W, WB);
    k_logm<<<((N_ - 1) * C_ + 255) / 256, 256, 0, stream>>>(masks, logm);
    k_table<<<(C_ + 255) / 256, 256, 0, stream>>>(cw, tidx, tcoef, tcnt);
    k_gemm<<<(N_ / 3) * (BT_ / 2), 32, 0, stream>>>(xb, WB, bias, nout);
    k_final<<<B_, 256, 0, stream>>>(gate, anc, nout, tidx, tcoef, tcnt, logm,
                                    output, sfmx);
}
